// EpisodeMultiheadAttentionBlock_23888608100570
// MI455X (gfx1250) — matrix operand order repaired and checked by reading, not compiled, not run on hardware
//
#include <hip/hip_runtime.h>
#include <math.h>

#define B_ 4
#define L_ 4096
#define E_ 256
#define Q_ 2048

typedef __attribute__((ext_vector_type(16))) __bf16 v16bf;
typedef __attribute__((ext_vector_type(8)))  __bf16 v8bf;
typedef __attribute__((ext_vector_type(8)))  float  v8f;

// scheduling hint helpers (mask: 0x8 = MFMA/WMMA, 0x20 = VMEM read)
#if __has_builtin(__builtin_amdgcn_sched_group_barrier)
#define SGB(mask, cnt) __builtin_amdgcn_sched_group_barrier(mask, cnt, 0)
#else
#define SGB(mask, cnt)
#endif
#if __has_builtin(__builtin_amdgcn_sched_barrier)
#define SBAR() __builtin_amdgcn_sched_barrier(0)
#else
#define SBAR()
#endif

// ---------------------------------------------------------------- helpers

__device__ __forceinline__ __bf16 f2bf(float f) {
  return (__bf16)f;   // native v_cvt on gfx1250
}

__device__ __forceinline__ v8f wmma_bf16(v16bf a, v16bf b, v8f c) {
  return __builtin_amdgcn_wmma_f32_16x16x32_bf16(false, a, false, b,
                                                 (short)0, c, false, false);
}

// A fragment (16x32 bf16, row-major src, ld elems).  ISA layout:
// lane l: row = l%16; elems 0..7 -> K = k0 + 8*(l/16) + j ; elems 8..15 -> +16
__device__ __forceinline__ v16bf load_a_bf(const __bf16* A, int ld, int k0) {
  int lane = threadIdx.x & 31;
  const __bf16* p = A + (long)(lane & 15) * ld + k0 + 8 * (lane >> 4);
  v8bf lo = *(const v8bf*)p;
  v8bf hi = *(const v8bf*)(p + 16);
  v16bf r;
#pragma unroll
  for (int i = 0; i < 8; ++i) { r[i] = lo[i]; r[i + 8] = hi[i]; }
  return r;
}

// A fragment built from an f32 row-major source (softmax weights)
__device__ __forceinline__ v16bf load_a_f32(const float* A, int ld, int k0) {
  int lane = threadIdx.x & 31;
  const float* p = A + (long)(lane & 15) * ld + k0 + 8 * (lane >> 4);
  const float4* p0 = (const float4*)p;
  const float4* p1 = (const float4*)(p + 16);
  float4 a0 = p0[0], a1 = p0[1], b0 = p1[0], b1 = p1[1];
  v16bf r;
  r[0]  = f2bf(a0.x); r[1]  = f2bf(a0.y); r[2]  = f2bf(a0.z); r[3]  = f2bf(a0.w);
  r[4]  = f2bf(a1.x); r[5]  = f2bf(a1.y); r[6]  = f2bf(a1.z); r[7]  = f2bf(a1.w);
  r[8]  = f2bf(b0.x); r[9]  = f2bf(b0.y); r[10] = f2bf(b0.z); r[11] = f2bf(b0.w);
  r[12] = f2bf(b1.x); r[13] = f2bf(b1.y); r[14] = f2bf(b1.z); r[15] = f2bf(b1.w);
  return r;
}

// B fragment (32x16).  Bt stored [N][K] row-major, pre-offset to column base.
// lane l: N = l%16 ; elems 0..7 -> K = k0 + 8*(l/16) + j ; elems 8..15 -> +16
__device__ __forceinline__ v16bf load_b_bf(const __bf16* Bt, long ld, int k0) {
  int lane = threadIdx.x & 31;
  const __bf16* p = Bt + (long)(lane & 15) * ld + k0 + 8 * (lane >> 4);
  v8bf lo = *(const v8bf*)p;
  v8bf hi = *(const v8bf*)(p + 16);
  v16bf r;
#pragma unroll
  for (int i = 0; i < 8; ++i) { r[i] = lo[i]; r[i + 8] = hi[i]; }
  return r;
}

// ---------------------------------------------------------------- cvt kernel

__global__ __launch_bounds__(256) void cvt_f32_bf16(const float* __restrict__ s,
                                                    __bf16* __restrict__ d, long n) {
  long i = ((long)blockIdx.x * blockDim.x + threadIdx.x) * 4;
  if (i < n) {
    float4 v = *(const float4*)(s + i);
    d[i + 0] = f2bf(v.x);
    d[i + 1] = f2bf(v.y);
    d[i + 2] = f2bf(v.z);
    d[i + 3] = f2bf(v.w);
  }
}

// ---------------------------------------------------------------- generic GEMM
// C[m,n] = A1[m,:] . B1t[n,:]  (+ A2[m,:] . B2t[n,:] if DUAL) + bias, + epilogue
// A rows live in a "mapped" space: mapped = (m/rowsPerB)*batchStride + rowOff + m%rowsPerB

enum { EP_ROPE = 0, EP_ROPESCALE, EP_VT, EP_BF, EP_SIG_RX, EP_SIG, EP_TANH_OUT };

template <int EPI, bool DUAL>
__global__ __launch_bounds__(128) void gemm16(
    const __bf16* __restrict__ A1, const __bf16* __restrict__ B1t,
    const float* __restrict__ bias1,
    const __bf16* __restrict__ A2, const __bf16* __restrict__ B2t,
    const float* __restrict__ bias2,
    float* __restrict__ outF, __bf16* __restrict__ outBF,
    __bf16* __restrict__ outT, long ldT,
    const int* __restrict__ key_index, const float* __restrict__ xF,
    const float* __restrict__ zF,
    int rowsPerB, int batchStride, int rowOff, int xStride, int xOff,
    float scale) {
  const int wave = threadIdx.x >> 5;
  const int lane = threadIdx.x & 31;
  const int half = lane >> 4;
  const int nl   = lane & 15;
  const int m0   = blockIdx.y * 16;
  const long mapped0  = (long)(m0 / rowsPerB) * batchStride + rowOff + (m0 % rowsPerB);
  const long xmapped0 = (long)(m0 / rowsPerB) * xStride   + xOff  + (m0 % rowsPerB);
  const int c0 = blockIdx.x * 128 + wave * 16;   // second tile at c0+64

  const __bf16* a1p  = A1 + mapped0 * E_;
  const __bf16* a2p  = DUAL ? (A2 + (long)m0 * E_) : nullptr;
  const __bf16* b1p0 = B1t + (long)c0 * E_;
  const __bf16* b1p1 = B1t + (long)(c0 + 64) * E_;
  const __bf16* b2p0 = DUAL ? (B2t + (long)c0 * E_) : nullptr;
  const __bf16* b2p1 = DUAL ? (B2t + (long)(c0 + 64) * E_) : nullptr;

  v8f acc0 = {0.f,0.f,0.f,0.f,0.f,0.f,0.f,0.f};
  v8f acc1 = {0.f,0.f,0.f,0.f,0.f,0.f,0.f,0.f};

  // software-pipelined K loop: fragments for step i+1 issued before WMMAs of i
  v16bf a_c  = load_a_bf(a1p, E_, 0);
  v16bf b0_c = load_b_bf(b1p0, E_, 0);
  v16bf b1_c = load_b_bf(b1p1, E_, 0);
  v16bf a2_c, c0_c, c1_c;
  if (DUAL) {
    a2_c = load_a_bf(a2p, E_, 0);
    c0_c = load_b_bf(b2p0, E_, 0);
    c1_c = load_b_bf(b2p1, E_, 0);
  }
#pragma unroll
  for (int i = 0; i < 8; ++i) {
    v16bf a_n, b0_n, b1_n, a2_n, c0_n, c1_n;
    if (i < 7) {
      int k0 = (i + 1) * 32;
      a_n  = load_a_bf(a1p, E_, k0);
      b0_n = load_b_bf(b1p0, E_, k0);
      b1_n = load_b_bf(b1p1, E_, k0);
      if (DUAL) {
        a2_n = load_a_bf(a2p, E_, k0);
        c0_n = load_b_bf(b2p0, E_, k0);
        c1_n = load_b_bf(b2p1, E_, k0);
      }
    }
    acc0 = wmma_bf16(a_c, b0_c, acc0);
    acc1 = wmma_bf16(a_c, b1_c, acc1);
    if (DUAL) {
      acc0 = wmma_bf16(a2_c, c0_c, acc0);
      acc1 = wmma_bf16(a2_c, c1_c, acc1);
    }
    a_c = a_n; b0_c = b0_n; b1_c = b1_n;
    if (DUAL) { a2_c = a2_n; c0_c = c0_n; c1_c = c1_n; }
  }
  // pin interleave: 2 K-steps of loads in flight ahead of each WMMA group
  if (DUAL) {
    SGB(0x20, 24);
#pragma unroll
    for (int g = 0; g < 6; ++g) { SGB(0x8, 4); SGB(0x20, 12); }
    SGB(0x8, 8);
  } else {
    SGB(0x20, 12);
#pragma unroll
    for (int g = 0; g < 6; ++g) { SGB(0x8, 2); SGB(0x20, 6); }
    SGB(0x8, 4);
  }
  SBAR();

#pragma unroll
  for (int t = 0; t < 2; ++t) {
    v8f acc = t ? acc1 : acc0;
    int n = (t ? c0 + 64 : c0) + nl;
    float bias = bias1[n] + (DUAL ? bias2[n] : 0.0f);
#pragma unroll
    for (int r = 0; r < 8; ++r) {
      int  M    = r + 8 * half;
      long m    = m0 + M;
      long mrow = mapped0 + M;
      float v   = acc[r] + bias;
      if (EPI == EP_ROPE || EPI == EP_ROPESCALE) {
        // inv_freq = 10000^(-(2i)/E), i = n/2 ; angle = idx * inv_freq
        float inv = __expf(-(float)(n & ~1) * (9.210340371976184f / 256.0f));
        float ang = (float)key_index[mrow] * inv;
        float c = __cosf(ang), s = __sinf(ang);
        float other = __shfl_xor(v, 1, 32);   // partner column (n^1)
        float res = ((n & 1) == 0) ? (v * c - other * s) : (other * s + v * c);
        if (EPI == EP_ROPESCALE) res *= scale;
        outBF[m * E_ + n] = f2bf(res);
      } else if (EPI == EP_VT) {
        outT[(long)n * ldT + mrow] = f2bf(v);            // transposed store
      } else if (EPI == EP_BF) {
        outBF[m * E_ + n] = f2bf(v);
      } else if (EPI == EP_SIG_RX) {
        float rg = 1.0f / (1.0f + __expf(-v));
        float x  = xF[(xmapped0 + M) * (long)E_ + n];
        outBF[m * E_ + n] = f2bf(rg * x);                // (r * x) in bf16
      } else if (EPI == EP_SIG) {
        outF[m * E_ + n] = 1.0f / (1.0f + __expf(-v));   // z
      } else {  // EP_TANH_OUT
        float h = tanhf(v);
        float z = zF[m * E_ + n];
        float x = xF[(xmapped0 + M) * (long)E_ + n];
        outF[m * E_ + n] = (1.0f - z) * x + z * h;       // final out
      }
    }
  }
}

// ---------------------------------------------------------------- attention
// One block (8 waves) per (batch, 16-query tile).  Scores -> w (global, L2
// resident) -> in-place softmax (w is a required output) -> attn = w @ v.

__global__ __launch_bounds__(256) void attn_kernel(
    const __bf16* __restrict__ q_bf, const __bf16* __restrict__ k_bf,
    const __bf16* __restrict__ vT, float* __restrict__ w_out,
    __bf16* __restrict__ attn_bf) {
  const int wave = threadIdx.x >> 5;
  const int lane = threadIdx.x & 31;
  const int half = lane >> 4;
  const int nl   = lane & 15;
  const int b    = blockIdx.x / (Q_ / 16);
  const int qt   = blockIdx.x % (Q_ / 16);
  const long qr0 = (long)b * Q_ + qt * 16;
  const int qpos0 = (L_ - Q_) + qt * 16;

  // hoist the 16x256 q tile as 8 A fragments (64 VGPRs)
  v16bf aq[8];
  {
    const __bf16* qp = q_bf + qr0 * E_;
#pragma unroll
    for (int i = 0; i < 8; ++i) aq[i] = load_a_bf(qp, E_, i * 32);
  }

  // -------- phase 1: masked raw scores, two key tiles per iteration
  for (int ktb = wave * 2; ktb < L_ / 16; ktb += 16) {
    int kbase = ktb * 16;                 // keys kbase..kbase+31
    v8f acc0 = {0.f,0.f,0.f,0.f,0.f,0.f,0.f,0.f};
    v8f acc1 = {0.f,0.f,0.f,0.f,0.f,0.f,0.f,0.f};
    if (kbase <= qpos0 + 15) {            // skip fully-masked tile pairs
      const __bf16* kp0 = k_bf + ((long)b * L_ + kbase) * E_;
      const __bf16* kp1 = kp0 + 16 * E_;
      v16bf b0c = load_b_bf(kp0, E_, 0);
      v16bf b1c = load_b_bf(kp1, E_, 0);
#pragma unroll
      for (int i = 0; i < 8; ++i) {
        v16bf b0n, b1n;
        if (i < 7) {
          b0n = load_b_bf(kp0, E_, (i + 1) * 32);
          b1n = load_b_bf(kp1, E_, (i + 1) * 32);
        }
        acc0 = wmma_bf16(aq[i], b0c, acc0);
        acc1 = wmma_bf16(aq[i], b1c, acc1);
        b0c = b0n; b1c = b1n;
      }
      // 2 K-steps of loads (8 reads) ahead, then [2 wmma][4 reads] pattern
      SGB(0x20, 8);
#pragma unroll
      for (int g = 0; g < 6; ++g) { SGB(0x8, 2); SGB(0x20, 4); }
      SGB(0x8, 4);
      SBAR();
    }
#pragma unroll
    for (int t = 0; t < 2; ++t) {
      v8f acc = t ? acc1 : acc0;
      int kpos = kbase + t * 16 + nl;
#pragma unroll
      for (int r = 0; r < 8; ++r) {
        int M = r + 8 * half;
        float v = (kpos <= qpos0 + M) ? acc[r] : -__builtin_inff();
        w_out[(qr0 + M) * (long)L_ + kpos] = v;
      }
    }
  }
  __threadfence();
  __syncthreads();

  // -------- phase 2: row softmax, normalized w written in place (output!)
  {
    int row = threadIdx.x >> 4;           // 16 rows x 16 threads
    int sub = threadIdx.x & 15;
    int qpos = qpos0 + row;
    float* wr = w_out + (qr0 + row) * (long)L_;
    float mx = -__builtin_inff();
    for (int j = sub; j <= qpos; j += 16) mx = fmaxf(mx, wr[j]);
#pragma unroll
    for (int o = 8; o >= 1; o >>= 1) mx = fmaxf(mx, __shfl_xor(mx, o, 32));
    float sum = 0.0f;
    for (int j = sub; j <= qpos; j += 16) {   // store exp, accumulate sum
      float e = __expf(wr[j] - mx);
      wr[j] = e;
      sum += e;
    }
#pragma unroll
    for (int o = 8; o >= 1; o >>= 1) sum += __shfl_xor(sum, o, 32);
    float inv = 1.0f / sum;
    for (int j = sub; j < L_; j += 16)        // normalize; zero masked tail
      wr[j] = (j <= qpos) ? wr[j] * inv : 0.0f;
  }
  __threadfence();
  __syncthreads();

  // -------- phase 3: attn_raw = w @ v ; one shared A fragment feeds two tiles
  {
    const float* wp = w_out + qr0 * (long)L_;
    int kend = (qpos0 + 16 + 31) & ~31;   // causal cap, rounded to K step
    int ct0 = wave, ct1 = wave + 8;       // 8 waves x 2 = 16 column tiles
    const __bf16* vp0 = vT + (long)(ct0 * 16) * (long)(B_ * L_) + (long)b * L_;
    const __bf16* vp1 = vT + (long)(ct1 * 16) * (long)(B_ * L_) + (long)b * L_;
    v8f acc0 = {0.f,0.f,0.f,0.f,0.f,0.f,0.f,0.f};
    v8f acc1 = {0.f,0.f,0.f,0.f,0.f,0.f,0.f,0.f};
    v16bf ac  = load_a_f32(wp, L_, 0);
    v16bf b0c = load_b_bf(vp0, B_ * L_, 0);
    v16bf b1c = load_b_bf(vp1, B_ * L_, 0);
    for (int kb = 0; kb < kend; kb += 32) {
      v16bf an, b0n, b1n;
      if (kb + 32 < kend) {
        an  = load_a_f32(wp, L_, kb + 32);
        b0n = load_b_bf(vp0, B_ * L_, kb + 32);
        b1n = load_b_bf(vp1, B_ * L_, kb + 32);
      }
      acc0 = wmma_bf16(ac, b0c, acc0);
      acc1 = wmma_bf16(ac, b1c, acc1);
      ac = an; b0c = b0n; b1c = b1n;
      SGB(0x20, 8);   // next step's 8 reads first ...
      SGB(0x8, 2);    // ... then this step's 2 WMMAs
    }
#pragma unroll
    for (int t = 0; t < 2; ++t) {
      v8f acc = t ? acc1 : acc0;
      int ct = t ? ct1 : ct0;
#pragma unroll
      for (int r = 0; r < 8; ++r) {
        int M = r + 8 * half;
        attn_bf[(qr0 + M) * (long)E_ + ct * 16 + nl] = f2bf(acc[r]);
      }
    }
  }
}

// ---------------------------------------------------------------- launcher

extern "C" void kernel_launch(void* const* d_in, const int* in_sizes, int n_in,
                              void* d_out, int out_size, void* d_ws, size_t ws_size,
                              hipStream_t stream) {
  (void)in_sizes; (void)n_in; (void)out_size; (void)ws_size;

  const float* key       = (const float*)d_in[0];
  const int*   key_index = (const int*)d_in[1];
  const float* W[10];
  const float* bia[10];
  for (int j = 0; j < 10; ++j) {              // q k v o xr yr xz yz xg yg
    W[j]   = (const float*)d_in[2 + 2 * j];
    bia[j] = (const float*)d_in[3 + 2 * j];
  }

  // workspace carve (all 256B aligned)
  char* p = (char*)d_ws;
  auto alloc = [&](size_t bytes) { void* r = p; p += (bytes + 255) & ~(size_t)255; return r; };
  __bf16* key_bf  = (__bf16*)alloc((size_t)B_ * L_ * E_ * 2);
  __bf16* q_bf    = (__bf16*)alloc((size_t)B_ * Q_ * E_ * 2);
  __bf16* k_bf    = (__bf16*)alloc((size_t)B_ * L_ * E_ * 2);
  __bf16* vT      = (__bf16*)alloc((size_t)E_ * B_ * L_ * 2);
  __bf16* y_bf    = (__bf16*)alloc((size_t)B_ * Q_ * E_ * 2);
  __bf16* attn_bf = (__bf16*)alloc((size_t)B_ * Q_ * E_ * 2);
  __bf16* rx_bf   = (__bf16*)alloc((size_t)B_ * Q_ * E_ * 2);
  float*  z_f     = (float*)alloc((size_t)B_ * Q_ * E_ * 4);
  __bf16* Wbf[10];
  for (int j = 0; j < 10; ++j) Wbf[j] = (__bf16*)alloc((size_t)E_ * E_ * 2);

  float* out_f = (float*)d_out;                       // (B,Q,E)
  float* w_f   = out_f + (long)B_ * Q_ * E_;          // (B,Q,L)

  // 1) bf16 conversions (4 elems/thread)
  {
    long n = (long)B_ * L_ * E_;
    cvt_f32_bf16<<<dim3((unsigned)((n / 4 + 255) / 256)), 256, 0, stream>>>(key, key_bf, n);
    for (int j = 0; j < 10; ++j)
      cvt_f32_bf16<<<dim3((E_ * E_ / 4 + 255) / 256), 256, 0, stream>>>(W[j], Wbf[j], E_ * E_);
  }

  const float scale = 0.0625f;                        // 1/sqrt(256)
  dim3 blk(128);
  // 2) q = RoPE(x_q @ Wq.T + bq) * scale   (rows mapped into key space)
  gemm16<EP_ROPESCALE, false><<<dim3(2, (B_ * Q_) / 16), blk, 0, stream>>>(
      key_bf, Wbf[0], bia[0], nullptr, nullptr, nullptr,
      nullptr, q_bf, nullptr, 0, key_index, nullptr, nullptr,
      Q_, L_, L_ - Q_, 0, 0, scale);
  // 3) k = RoPE(key @ Wk.T + bk)
  gemm16<EP_ROPE, false><<<dim3(2, (B_ * L_) / 16), blk, 0, stream>>>(
      key_bf, Wbf[1], bia[1], nullptr, nullptr, nullptr,
      nullptr, k_bf, nullptr, 0, key_index, nullptr, nullptr,
      L_, L_, 0, 0, 0, 1.0f);
  // 4) v = key @ Wv.T + bv  (stored transposed for the w@v B operand)
  gemm16<EP_VT, false><<<dim3(2, (B_ * L_) / 16), blk, 0, stream>>>(
      key_bf, Wbf[2], bia[2], nullptr, nullptr, nullptr,
      nullptr, nullptr, vT, (long)B_ * L_, key_index, nullptr, nullptr,
      L_, L_, 0, 0, 0, 1.0f);
  // 5) attention: scores -> softmax (writes w output) -> attn_raw
  attn_kernel<<<dim3(B_ * (Q_ / 16)), 256, 0, stream>>>(q_bf, k_bf, vT, w_f, attn_bf);
  // 6) y = attn_raw @ Wo.T + bo
  gemm16<EP_BF, false><<<dim3(2, (B_ * Q_) / 16), blk, 0, stream>>>(
      attn_bf, Wbf[3], bia[3], nullptr, nullptr, nullptr,
      nullptr, y_bf, nullptr, 0, key_index, nullptr, nullptr,
      Q_, Q_, 0, 0, 0, 1.0f);
  // 7) r = sigmoid(x@Wxr.T + y@Wyr.T + b); store rx = r*x (bf16)
  gemm16<EP_SIG_RX, true><<<dim3(2, (B_ * Q_) / 16), blk, 0, stream>>>(
      key_bf, Wbf[4], bia[4], y_bf, Wbf[5], bia[5],
      nullptr, rx_bf, nullptr, 0, key_index, key, nullptr,
      Q_, L_, L_ - Q_, L_, L_ - Q_, 1.0f);
  // 8) z = sigmoid(x@Wxz.T + y@Wyz.T + b)
  gemm16<EP_SIG, true><<<dim3(2, (B_ * Q_) / 16), blk, 0, stream>>>(
      key_bf, Wbf[6], bia[6], y_bf, Wbf[7], bia[7],
      z_f, nullptr, nullptr, 0, key_index, nullptr, nullptr,
      Q_, L_, L_ - Q_, 0, 0, 1.0f);
  // 9) h = tanh(rx@Wxg.T + y@Wyg.T + b); out = (1-z)*x + z*h
  gemm16<EP_TANH_OUT, true><<<dim3(2, (B_ * Q_) / 16), blk, 0, stream>>>(
      rx_bf, Wbf[8], bia[8], y_bf, Wbf[9], bia[9],
      out_f, nullptr, nullptr, 0, key_index, key, z_f,
      Q_, Q_, 0, L_, L_ - Q_, 1.0f);
}
